// UnionLayer_36301063585948
// MI455X (gfx1250) — compile-verified, matrix operand order checked by reading, not run on hardware
//
#include <hip/hip_runtime.h>

// UnionLayer (fuzzy conjunction/disjunction) on MI455X / gfx1250.
//
// Math:  sum_d log(1 - u*w) = -S,  S = sum_{d,k} (u^k / k) * (w^k), k = 1..16
//        con = 1/(1+S_con),  dis = S_dis/(1+S_dis)
// The (d,k)-expanded reduction is a plain GEMM -> v_wmma_f32_16x16x32_f16,
// with power fragments generated in-register via packed f16 multiplies.
//
// Wave tile 32x64 (2 A-frags x 4 B-frags): 64 pk_mul per 8 WMMA = 8 VALU/WMMA,
// co-executing under the matrix pipe (16-bit WMMA issues on TRANS pipe).

typedef __attribute__((ext_vector_type(16))) _Float16 v16h;
typedef __attribute__((ext_vector_type(8)))  float    v8f;
typedef __attribute__((ext_vector_type(4)))  float    v4f;

namespace {

constexpr int kB       = 1024;  // batch
constexpr int kD       = 512;   // reduction dim
constexpr int kN       = 256;   // units per side (out has 2*kN cols)
constexpr int kOutCols = 2 * kN;
constexpr int KPOW     = 16;    // Mercator series terms (z <= 0.5 -> err < 1e-6)

constexpr int MA = 2;  // 16-row A tiles per wave   (wave tile M = 32)
constexpr int NB = 4;  // 16-col B tiles per wave   (wave tile N = 64)

__device__ __forceinline__ v16h splat16(_Float16 s) {
  v16h r;
#pragma unroll
  for (int e = 0; e < 16; ++e) r[e] = s;
  return r;
}

// Load one lane's 16 f16 fragment elements: two contiguous 8-float runs at
// base[0..7] and base[16..23] (16-bit A/B WMMA VGPR layout: a lane holds
// K = {kb..kb+7, 16+kb..16+kb+7}, kb = 8*(lane>>4)).  u = fma(scale, x, offs)
// implements u = x (dis) or u = 1 - x (con) before f32->f16 conversion.
__device__ __forceinline__ v16h load_frag(const float* __restrict__ base,
                                          float scale, float offs) {
  v4f f0 = *(const v4f*)(base + 0);
  v4f f1 = *(const v4f*)(base + 4);
  v4f f2 = *(const v4f*)(base + 16);
  v4f f3 = *(const v4f*)(base + 20);
  v16h t;
#pragma unroll
  for (int e = 0; e < 4; ++e) {
    t[e + 0]  = (_Float16)fmaf(scale, f0[e], offs);
    t[e + 4]  = (_Float16)fmaf(scale, f1[e], offs);
    t[e + 8]  = (_Float16)fmaf(scale, f2[e], offs);
    t[e + 12] = (_Float16)fmaf(scale, f3[e], offs);
  }
  return t;
}

}  // namespace

// 4 waves (wave32) per block, arranged 2 (M) x 2 (N); wave tile = 32x64,
// block tile = 64x128 over the (1024 x 512) output.
__global__ __launch_bounds__(128) void union_series_wmma(
    const float* __restrict__ x,     // (B, D)
    const float* __restrict__ Wcon,  // (N, D)
    const float* __restrict__ Wdis,  // (N, D)
    float* __restrict__ out) {       // (B, 2N)
  const int lane  = threadIdx.x & 31;
  const int wave  = threadIdx.x >> 5;
  const int waveM = wave >> 1;
  const int waveN = wave & 1;
  const int half  = lane >> 4;   // 0: K 0..7 / 16..23, 1: K 8..15 / 24..31
  const int lm    = lane & 15;

  const int rowBase = blockIdx.x * 64 + waveM * 32;        // b base of wave tile
  const int colGlob = blockIdx.y * 128 + waveN * 64;       // global out col base
  const bool isDis  = (colGlob >= kN);                     // 64-col tiles never straddle
  const int  n0     = colGlob - (isDis ? kN : 0);
  const float* __restrict__ W = isDis ? Wdis : Wcon;
  const float uScale = isDis ? 1.0f : -1.0f;   // u = x  or  u = 1 - x
  const float uOffs  = isDis ? 0.0f : 1.0f;

  v8f acc[MA][NB] = {};  // S accumulators (f32)

  for (int d0 = 0; d0 < kD; d0 += 32) {
    // Base (power-1) fragments for this 32-wide d chunk.
    v16h ua[MA], wb[NB];
#pragma unroll
    for (int i = 0; i < MA; ++i)
      ua[i] = load_frag(x + (size_t)(rowBase + i * 16 + lm) * kD + d0 + half * 8,
                        uScale, uOffs);
#pragma unroll
    for (int j = 0; j < NB; ++j)
      wb[j] = load_frag(W + (size_t)(n0 + j * 16 + lm) * kD + d0 + half * 8,
                        1.0f, 0.0f);

    // ap holds u^k / k (1/k coefficient folded into the A side, which is
    // shared by all NB B-tiles), bp holds w^k.
    v16h ap[MA], bp[NB];
#pragma unroll
    for (int i = 0; i < MA; ++i) ap[i] = ua[i];
#pragma unroll
    for (int j = 0; j < NB; ++j) bp[j] = wb[j];

#pragma unroll
    for (int k = 1; k <= KPOW; ++k) {
#pragma unroll
      for (int i = 0; i < MA; ++i)
#pragma unroll
        for (int j = 0; j < NB; ++j)
          acc[i][j] = __builtin_amdgcn_wmma_f32_16x16x32_f16(
              false, ap[i], false, bp[j], (short)0, acc[i][j], false, false);
      if (k < KPOW) {
        // u^{k+1}/(k+1) = (u^k/k) * u * (k/(k+1));   w^{k+1} = w^k * w
        const v16h s = splat16((_Float16)((float)k / (float)(k + 1)));
#pragma unroll
        for (int i = 0; i < MA; ++i) ap[i] = ap[i] * ua[i] * s;
#pragma unroll
        for (int j = 0; j < NB; ++j) bp[j] = bp[j] * wb[j];
      }
    }
  }

  // Epilogue: C/D layout -> VGPR e holds M = e + 8*half, N = lane&15.
#pragma unroll
  for (int i = 0; i < MA; ++i) {
#pragma unroll
    for (int j = 0; j < NB; ++j) {
#pragma unroll
      for (int e = 0; e < 8; ++e) {
        const int row = rowBase + i * 16 + half * 8 + e;
        const int col = colGlob + j * 16 + lm;
        const float S   = acc[i][j][e];
        const float inv = 1.0f / (1.0f + S);  // S >= 0 always
        out[(size_t)row * kOutCols + col] = isDis ? (S * inv) : inv;
      }
    }
  }
}

extern "C" void kernel_launch(void* const* d_in, const int* in_sizes, int n_in,
                              void* d_out, int out_size, void* d_ws,
                              size_t ws_size, hipStream_t stream) {
  (void)in_sizes; (void)n_in; (void)out_size; (void)d_ws; (void)ws_size;
  const float* x    = (const float*)d_in[0];
  const float* Wcon = (const float*)d_in[1];
  const float* Wdis = (const float*)d_in[2];
  float* out = (float*)d_out;

  dim3 grid(kB / 64, kOutCols / 128);  // (16, 4)
  dim3 block(128);                     // 4 wave32s
  union_series_wmma<<<grid, block, 0, stream>>>(x, Wcon, Wdis, out);
}
